// MaskFormerDecoderGraph_74663711474355
// MI455X (gfx1250) — compile-verified, matrix operand order checked by reading, not run on hardware
//
#include <hip/hip_runtime.h>

// ---------------- problem constants ----------------
#define GH 192
#define GW 192
#define GN (GH * GW)        // 36864 nodes per batch
#define GB 4
#define GC 256
#define BN (GB * GN)        // 147456 rows
#define INF_ 768            // C*(K+1)
#define HID_ 256
#define INV_2SIG2 78.125f   // 1 / (2*0.08^2)

typedef float v8f __attribute__((ext_vector_type(8)));
typedef __bf16 v16bf __attribute__((ext_vector_type(16)));

static __device__ __forceinline__ unsigned short f2bf(float f) {
  union { float f; unsigned u; } x; x.f = f;
  unsigned r = x.u + 0x7FFFu + ((x.u >> 16) & 1u);   // round-to-nearest-even
  return (unsigned short)(r >> 16);
}

// ---------------- 1: per-batch min/max of rgt map ----------------
__global__ __launch_bounds__(256) void k_rgt_minmax(const float* __restrict__ rgt,
                                                    float* __restrict__ mm) {
  __shared__ float smin[256], smax[256];
  const int b = blockIdx.x, tid = threadIdx.x;
  const float* r = rgt + (size_t)b * GN;
  float mn = 3.4e38f, mx = -3.4e38f;
  for (int i = tid; i < GN; i += 256) { float v = r[i]; mn = fminf(mn, v); mx = fmaxf(mx, v); }
  smin[tid] = mn; smax[tid] = mx; __syncthreads();
  for (int s = 128; s > 0; s >>= 1) {
    if (tid < s) { smin[tid] = fminf(smin[tid], smin[tid + s]); smax[tid] = fmaxf(smax[tid], smax[tid + s]); }
    __syncthreads();
  }
  if (tid == 0) { mm[b * 2] = smin[0]; mm[b * 2 + 1] = smax[0]; }
}

// ---------------- 2: implicit edge weights + degree tables ----------------
__global__ __launch_bounds__(256) void k_edge_weights(const float* __restrict__ rgt,
                                                      const float* __restrict__ mm,
                                                      float* __restrict__ wE, float* __restrict__ wS,
                                                      float* __restrict__ dinv, float* __restrict__ dinvl) {
  const int n = blockIdx.x * 256 + threadIdx.x;
  const int b = blockIdx.y;
  const float rmin = mm[b * 2], rmax = mm[b * 2 + 1];
  const float inv = 1.0f / fmaxf(rmax - rmin, 1e-12f);
  const float* r = rgt + (size_t)b * GN;
  const int col = n % GW, row = n / GW;
  const float rc = (r[n] - rmin) * inv;
  float we = 0.f, ws = 0.f, deg = 0.f;
  if (col < GW - 1) { float d = rc - (r[n + 1]  - rmin) * inv; we = __expf(-(d * d) * INV_2SIG2); deg += we; }
  if (row < GH - 1) { float d = rc - (r[n + GW] - rmin) * inv; ws = __expf(-(d * d) * INV_2SIG2); deg += ws; }
  if (col > 0)      { float d = rc - (r[n - 1]  - rmin) * inv; deg += __expf(-(d * d) * INV_2SIG2); }
  if (row > 0)      { float d = rc - (r[n - GW] - rmin) * inv; deg += __expf(-(d * d) * INV_2SIG2); }
  const size_t bn = (size_t)b * GN + n;
  wE[bn] = we; wS[bn] = ws;
  dinv[bn]  = rsqrtf(deg + 1e-12f);  // diffusion norm
  dinvl[bn] = rsqrtf(deg + 1.0f);    // GCN norm with self loop
}

// ---------------- 3/4: one diffusion hop (5-pt weighted gather, (C,N) layout) ----------------
__global__ __launch_bounds__(256) void k_diffuse_hop(const float* __restrict__ src,
                                                     float* __restrict__ dst,
                                                     const float* __restrict__ wE, const float* __restrict__ wS,
                                                     const float* __restrict__ dinv) {
  const int n = blockIdx.x * 256 + threadIdx.x;
  const int c = blockIdx.y, b = blockIdx.z;
  const size_t bn = (size_t)b * GN + n;
  const float* x = src + ((size_t)b * GC + c) * GN;
  const int col = n % GW, row = n / GW;
  float acc = 0.f;
  if (col < GW - 1) acc += wE[bn]      * dinv[bn + 1]  * x[n + 1];
  if (col > 0)      acc += wE[bn - 1]  * dinv[bn - 1]  * x[n - 1];
  if (row < GH - 1) acc += wS[bn]      * dinv[bn + GW] * x[n + GW];
  if (row > 0)      acc += wS[bn - GW] * dinv[bn - GW] * x[n - GW];
  dst[((size_t)b * GC + c) * GN + n] = acc * dinv[bn];
}

// ---------------- 5: transpose (C,N)->(N,C) + bf16 pack into x_all (BN x 768) ----------------
__global__ __launch_bounds__(256) void k_pack_xall(const float* __restrict__ x0,
                                                   const float* __restrict__ h1,
                                                   const float* __restrict__ h2,
                                                   unsigned short* __restrict__ xall) {
  __shared__ float t[64][65];
  const int n0 = blockIdx.x * 64;
  const int s  = blockIdx.y >> 2;
  const int c0 = (blockIdx.y & 3) * 64;
  const int b  = blockIdx.z;
  const float* src = (s == 0) ? x0 : ((s == 1) ? h1 : h2);
  const float* base = src + ((size_t)b * GC + c0) * GN + n0;
  const int tid = threadIdx.x;
#pragma unroll
  for (int i = 0; i < 16; ++i) {
    int idx = i * 256 + tid; int cr = idx >> 6, nc = idx & 63;
    t[cr][nc] = base[(size_t)cr * GN + nc];
  }
  __syncthreads();
#pragma unroll
  for (int i = 0; i < 16; ++i) {
    int idx = i * 256 + tid; int nr = idx >> 6, cc = idx & 63;
    xall[((size_t)b * GN + n0 + nr) * INF_ + s * GC + c0 + cc] = f2bf(t[cc][nr]);
  }
}

// ---------------- 6/7: swizzle weights into WMMA 16-bit B-fragment layout ----------------
// tile (kt,nt): lane l holds col n=nt*16+(l&15), half h=l>>4; vgpr v packs K-pair:
//   v<4 : k0 = kt*32 + h*8 + 2v        v>=4 : k0 = kt*32 + 16 + h*8 + 2(v-4)
__global__ __launch_bounds__(256) void k_swizzle_w(const float* __restrict__ Wm,
                                                   unsigned* __restrict__ swz, int Kdim) {
  const int e = blockIdx.x * 256 + threadIdx.x;
  if (e >= (Kdim / 32) * 4096) return;
  const int v = e & 7, l = (e >> 3) & 31, nt = (e >> 8) & 15, kt = e >> 12;
  const int n = nt * 16 + (l & 15), h = l >> 4;
  const int k0 = kt * 32 + ((v < 4) ? (h * 8 + v * 2) : (16 + h * 8 + (v - 4) * 2));
  const unsigned lo = f2bf(Wm[(size_t)k0 * 256 + n]);
  const unsigned hi = f2bf(Wm[(size_t)(k0 + 1) * 256 + n]);
  swz[e] = lo | (hi << 16);
}

// ---------------- TDM: DMA one contiguous 16xK bf16 A-tile into LDS ----------------
// D# per CDNA5 ISA: group0 = count/lds_addr/global_addr/type,
// group1 = data_size + dims/strides (single row of 8-byte units), groups 2/3 zeroed.
template <int KDIM>
static __device__ __forceinline__ void tdm_stage_tile(const unsigned short* gptr,
                                                      unsigned lds_off) {
  typedef unsigned v4u __attribute__((ext_vector_type(4)));
  typedef int v4i __attribute__((ext_vector_type(4)));
  typedef int v8i __attribute__((ext_vector_type(8)));
  const unsigned units = (unsigned)((16 * KDIM * 2) >> 3);   // tile bytes in 8B units
  const unsigned long long ga = (unsigned long long)gptr;
  v4u g0;
  g0[0] = 1u;                                                // count=1 (valid user D#)
  g0[1] = lds_off;                                           // lds_addr (bytes)
  g0[2] = (unsigned)ga;                                      // global_addr[31:0]
  g0[3] = (unsigned)((ga >> 32) & 0x01FFFFFFu) | 0x80000000u;// global_addr[56:32] | type=2
  v8i g1;
  g1[0] = 0x00030000;                                        // data_size=8B, wg_mask=0
  g1[1] = (int)((units & 0xFFFFu) << 16);                    // tensor_dim0[15:0]
  g1[2] = (int)((units >> 16) | (1u << 16));                 // tensor_dim0[31:16] | tensor_dim1=1
  g1[3] = (int)((units & 0xFFFFu) << 16);                    // tile_dim0
  g1[4] = 1;                                                 // tile_dim1=1, tile_dim2=0
  g1[5] = (int)units;                                        // tensor_dim0_stride[31:0]
  g1[6] = 0;                                                 // stride hi | tensor_dim1_stride lo
  g1[7] = 0;
  v4i g2 = {0, 0, 0, 0}, g3 = {0, 0, 0, 0};
  asm volatile("tensor_load_to_lds %0, %1, %2, %3"
               :: "s"(g0), "s"(g1), "s"(g2), "s"(g3)
               : "memory");
}

// ---------------- 8/10: bf16 WMMA GEMM  Out(BNx256) = A(BNxK) @ Wswz ----------------
template <int KDIM>
__global__ __launch_bounds__(256) void k_gemm_bf16_wmma(const unsigned short* __restrict__ A,
                                                        const unsigned* __restrict__ Bswz,
                                                        float* __restrict__ Out) {
  __shared__ __align__(16) unsigned sA[16 * (KDIM / 2)];   // 16 rows x K bf16, packed pairs
  const int tid = threadIdx.x;
  const int rowTile = blockIdx.x;
  // wave 0 DMAs the A-tile via the Tensor Data Mover, others wait at the barrier
  if (tid < 32) {
    tdm_stage_tile<KDIM>(A + (size_t)rowTile * 16 * KDIM,
                         (unsigned)(unsigned long long)(void*)sA);
    __builtin_amdgcn_s_wait_tensorcnt(0);
  }
  __syncthreads();

  const int lane = tid & 31, wv = tid >> 5;
  const int colTile = blockIdx.y * 8 + wv;
  const int half = lane >> 4, mloc = lane & 15;
  union { unsigned u[8]; v16bf v; } a, bfr;
  v8f acc = {};
  const int KT = KDIM / 32;
  for (int kt = 0; kt < KT; ++kt) {
    // A fragment from LDS (16-bit A 16x32 VGPR layout)
#pragma unroll
    for (int v = 0; v < 8; ++v) {
      const int k0 = kt * 32 + ((v < 4) ? (half * 8 + v * 2) : (16 + half * 8 + (v - 4) * 2));
      a.u[v] = sA[mloc * (KDIM / 2) + (k0 >> 1)];
    }
    // B fragment: 8 consecutive dwords per lane (pre-swizzled)
    const uint4* bp = (const uint4*)(Bswz + ((((size_t)kt * 16 + colTile) * 32 + lane) << 3));
    if (kt + 1 < KT)
      __builtin_prefetch((const void*)(Bswz + ((((size_t)(kt + 1) * 16 + colTile) * 32 + lane) << 3)), 0, 1);
    const uint4 b0 = bp[0], b1 = bp[1];
    bfr.u[0] = b0.x; bfr.u[1] = b0.y; bfr.u[2] = b0.z; bfr.u[3] = b0.w;
    bfr.u[4] = b1.x; bfr.u[5] = b1.y; bfr.u[6] = b1.z; bfr.u[7] = b1.w;
    acc = __builtin_amdgcn_wmma_f32_16x16x32_bf16(false, a.v, false, bfr.v,
                                                  (short)0, acc, false, false);
  }
  // C/D layout: lane holds col n=lane&15, rows m = 8*(lane>>4) + r
  float* O = Out + (size_t)rowTile * 16 * 256 + colTile * 16;
  const int ncol = lane & 15, m0 = (lane >> 4) * 8;
#pragma unroll
  for (int r = 0; r < 8; ++r) O[(size_t)(m0 + r) * 256 + ncol] = acc[r];
}

// ---------------- 9: GCN aggregate + bias + LayerNorm + ReLU -> bf16 ----------------
__global__ __launch_bounds__(256) void k_agg_ln_relu(const float* __restrict__ Hin,
                                                     const float* __restrict__ bias,
                                                     const float* __restrict__ gamma,
                                                     const float* __restrict__ beta,
                                                     const float* __restrict__ wE, const float* __restrict__ wS,
                                                     const float* __restrict__ dinvl,
                                                     unsigned short* __restrict__ A2) {
  __shared__ float red[256];
  const int blk = blockIdx.x;
  const int b = blk / GN, n = blk % GN;
  const size_t bn = (size_t)b * GN + n;
  const int c = threadIdx.x;
  const int col = n % GW, row = n / GW;
  const float dl = dinvl[bn];
  float acc = Hin[bn * 256 + c] * (dl * dl);                    // self loop (weight 1)
  if (col < GW - 1) acc += wE[bn]      * dl * dinvl[bn + 1]  * Hin[(bn + 1)  * 256 + c];
  if (col > 0)      acc += wE[bn - 1]  * dl * dinvl[bn - 1]  * Hin[(bn - 1)  * 256 + c];
  if (row < GH - 1) acc += wS[bn]      * dl * dinvl[bn + GW] * Hin[(bn + GW) * 256 + c];
  if (row > 0)      acc += wS[bn - GW] * dl * dinvl[bn - GW] * Hin[(bn - GW) * 256 + c];
  acc += bias[c];
  // LayerNorm over 256 channels
  red[c] = acc; __syncthreads();
  for (int s = 128; s > 0; s >>= 1) { if (c < s) red[c] += red[c + s]; __syncthreads(); }
  const float mu = red[0] * (1.0f / 256.0f); __syncthreads();
  const float d = acc - mu;
  red[c] = d * d; __syncthreads();
  for (int s = 128; s > 0; s >>= 1) { if (c < s) red[c] += red[c + s]; __syncthreads(); }
  const float var = red[0] * (1.0f / 256.0f);
  float y = d * rsqrtf(var + 1e-5f) * gamma[c] + beta[c];
  y = fmaxf(y, 0.f);
  A2[bn * 256 + c] = f2bf(y);
}

// ---------------- 11: GCN aggregate + bias (f32 out) ----------------
__global__ __launch_bounds__(256) void k_agg2(const float* __restrict__ Hin,
                                              const float* __restrict__ bias,
                                              const float* __restrict__ wE, const float* __restrict__ wS,
                                              const float* __restrict__ dinvl,
                                              float* __restrict__ Yout) {
  const int blk = blockIdx.x;
  const int b = blk / GN, n = blk % GN;
  const size_t bn = (size_t)b * GN + n;
  const int c = threadIdx.x;
  const int col = n % GW, row = n / GW;
  const float dl = dinvl[bn];
  float acc = Hin[bn * 256 + c] * (dl * dl);
  if (col < GW - 1) acc += wE[bn]      * dl * dinvl[bn + 1]  * Hin[(bn + 1)  * 256 + c];
  if (col > 0)      acc += wE[bn - 1]  * dl * dinvl[bn - 1]  * Hin[(bn - 1)  * 256 + c];
  if (row < GH - 1) acc += wS[bn]      * dl * dinvl[bn + GW] * Hin[(bn + GW) * 256 + c];
  if (row > 0)      acc += wS[bn - GW] * dl * dinvl[bn - GW] * Hin[(bn - GW) * 256 + c];
  Yout[bn * 256 + c] = acc + bias[c];
}

// ---------------- 12: transpose (N,C)->(C,N) + gated residual ----------------
__global__ __launch_bounds__(256) void k_final(const float* __restrict__ y,
                                               const float* __restrict__ feat,
                                               const float* __restrict__ gate,
                                               float* __restrict__ out) {
  __shared__ float t[64][65];
  const int n0 = blockIdx.x * 64, c0 = blockIdx.y * 64, b = blockIdx.z;
  const int tid = threadIdx.x;
#pragma unroll
  for (int i = 0; i < 16; ++i) {
    int idx = i * 256 + tid; int nr = idx >> 6, cc = idx & 63;
    t[cc][nr] = y[((size_t)b * GN + n0 + nr) * 256 + c0 + cc];
  }
  __syncthreads();
  const float g = fminf(fmaxf(gate[0], 0.f), 1.f);
#pragma unroll
  for (int i = 0; i < 16; ++i) {
    int idx = i * 256 + tid; int cr = idx >> 6, nc = idx & 63;
    const size_t o = ((size_t)b * GC + c0 + cr) * GN + n0 + nc;
    const float f = feat[o];
    out[o] = f + g * (t[cr][nc] - f);
  }
}

// ---------------- host side ----------------
extern "C" void kernel_launch(void* const* d_in, const int* in_sizes, int n_in,
                              void* d_out, int out_size, void* d_ws, size_t ws_size,
                              hipStream_t stream) {
  const float* feat  = (const float*)d_in[0];   // (B,C,H,W)
  const float* rgt   = (const float*)d_in[1];   // (B,1,H,W)
  const float* W1    = (const float*)d_in[2];   // (768,256)
  const float* b1    = (const float*)d_in[3];
  const float* gamma = (const float*)d_in[4];
  const float* beta  = (const float*)d_in[5];
  const float* W2    = (const float*)d_in[6];   // (256,256)
  const float* b2    = (const float*)d_in[7];
  const float* gate  = (const float*)d_in[8];
  float* out = (float*)d_out;

  char* ws = (char*)d_ws;
  size_t off = 0;
  auto take = [&](size_t bytes) { size_t r = off; off = (off + bytes + 255) & ~(size_t)255; return r; };
  const size_t NBf = (size_t)GB * GN * sizeof(float);
  float* mm     = (float*)(ws + take(64));
  float* wE     = (float*)(ws + take(NBf));
  float* wS     = (float*)(ws + take(NBf));
  float* dinv   = (float*)(ws + take(NBf));
  float* dinvl  = (float*)(ws + take(NBf));
  unsigned* w1s = (unsigned*)(ws + take((size_t)(INF_ / 32) * 4096 * 4));
  unsigned* w2s = (unsigned*)(ws + take((size_t)(HID_ / 32) * 4096 * 4));
  const size_t planeF = (size_t)GB * GC * GN * sizeof(float);   // 151 MB
  char* h1region   = ws + take(planeF);                          // h1 -> later H (GEMM1 out) -> later y2agg
  char* h2region   = ws + take(planeF);                          // h2 -> later A2 (bf16)
  char* xallregion = ws + take((size_t)BN * INF_ * 2);           // x_all -> later H2 (GEMM2 out)

  float* h1 = (float*)h1region;
  float* h2 = (float*)h2region;
  unsigned short* xall = (unsigned short*)xallregion;
  float* Hbuf = (float*)h1region;                   // GEMM1 out (aliases h1, dead then)
  unsigned short* A2 = (unsigned short*)h2region;   // LN/ReLU out (aliases h2)
  float* H2f = (float*)xallregion;                  // GEMM2 out (aliases x_all)
  float* y2 = (float*)h1region;                     // agg2 out (aliases Hbuf)

  // 1. per-batch min/max
  k_rgt_minmax<<<GB, 256, 0, stream>>>(rgt, mm);
  // 2. edge weights + degree tables
  k_edge_weights<<<dim3(GN / 256, GB), 256, 0, stream>>>(rgt, mm, wE, wS, dinv, dinvl);
  // 3/4. diffusion hops (feat is already (B,C,N))
  k_diffuse_hop<<<dim3(GN / 256, GC, GB), 256, 0, stream>>>(feat, h1, wE, wS, dinv);
  k_diffuse_hop<<<dim3(GN / 256, GC, GB), 256, 0, stream>>>(h1, h2, wE, wS, dinv);
  // 5. pack x_all = [x | h1 | h2] as bf16 (N,3C)
  k_pack_xall<<<dim3(GN / 64, 12, GB), 256, 0, stream>>>(feat, h1, h2, xall);
  // 6/7. weight swizzles
  k_swizzle_w<<<(INF_ / 32) * 4096 / 256, 256, 0, stream>>>(W1, w1s, INF_);
  k_swizzle_w<<<(HID_ / 32) * 4096 / 256, 256, 0, stream>>>(W2, w2s, HID_);
  // 8. GEMM1 (bf16 WMMA, TDM-staged A): H = x_all @ W1
  k_gemm_bf16_wmma<INF_><<<dim3(BN / 16, 2), 256, 0, stream>>>(xall, w1s, Hbuf);
  // 9. aggregate + bias + LN + ReLU -> bf16
  k_agg_ln_relu<<<BN, 256, 0, stream>>>(Hbuf, b1, gamma, beta, wE, wS, dinvl, A2);
  // 10. GEMM2: H2 = A2 @ W2
  k_gemm_bf16_wmma<HID_><<<dim3(BN / 16, 2), 256, 0, stream>>>(A2, w2s, H2f);
  // 11. aggregate + bias
  k_agg2<<<BN, 256, 0, stream>>>(H2f, b2, wE, wS, dinvl, y2);
  // 12. transpose + gated residual
  k_final<<<dim3(GN / 64, GC / 64, GB), 256, 0, stream>>>(y2, feat, gate, out);

  (void)in_sizes; (void)n_in; (void)out_size; (void)ws_size;
}
